// TransformerBlock_34265249087540
// MI455X (gfx1250) — matrix operand order repaired and checked by reading, not compiled, not run on hardware
//
#include <hip/hip_runtime.h>
#include <math.h>

// ---------------------------------------------------------------------------
// Types for CDNA5 WMMA (wave32)
// ---------------------------------------------------------------------------
typedef __bf16 bf16_t;
typedef __attribute__((ext_vector_type(16))) __bf16 v16bf;
typedef __attribute__((ext_vector_type(8)))  float  v8f;
typedef __attribute__((ext_vector_type(8)))  unsigned short ushort8;
typedef __attribute__((ext_vector_type(4)))  unsigned int   uint4v;
typedef __attribute__((ext_vector_type(4)))  unsigned int   uint32x4;
typedef __attribute__((ext_vector_type(8)))  unsigned int   uint32x8;

union FragU {
    v16bf v;
    ushort8 h[2];
};

__device__ __forceinline__ unsigned short f2bf(float x) {
    union { float f; unsigned int u; } c;
    c.f = x;
    unsigned int u = c.u;
    unsigned int r = u + 0x7FFFu + ((u >> 16) & 1u);   // round-to-nearest-even
    return (unsigned short)(r >> 16);
}

__device__ __forceinline__ v8f wmma_bf16(v16bf a, v16bf b, v8f c) {
    // D = A(16x32 bf16) x B(32x16 bf16) + C(16x16 f32)
    return __builtin_amdgcn_wmma_f32_16x16x32_bf16(
        /*neg_a=*/false, a, /*neg_b=*/false, b,
        /*c_mod=*/(short)0, c, /*reuse_a=*/false, /*reuse_b=*/false);
}

// A-fragment (16x32, MxK): lane L -> row mBase+(L&15); element e<8 -> k=kBase+8*half+e,
// e>=8 -> k=kBase+16+8*half+(e-8). Two contiguous 16B chunks.
__device__ __forceinline__ v16bf load_afrag(const unsigned short* lds, int rowStride,
                                            int mBase, int kBase) {
    const int lane = threadIdx.x & 31;
    const int m = mBase + (lane & 15);
    const int half = lane >> 4;
    const unsigned short* p = lds + m * rowStride + kBase + 8 * half;
    FragU f;
    f.h[0] = *(const ushort8*)p;
    f.h[1] = *(const ushort8*)(p + 16);
    return f.v;
}

// B-fragment (32x16, KxN) from LDS stored as rows-of-N with contiguous K:
// lane L -> col nBase+(L&15); e<8 -> k=kBase+8*half+e, e>=8 -> k=kBase+16+8*half+(e-8).
__device__ __forceinline__ v16bf load_bfrag(const unsigned short* lds, int rowStride,
                                            int nBase, int kBase) {
    const int lane = threadIdx.x & 31;
    const int n = nBase + (lane & 15);
    const int half = lane >> 4;
    const unsigned short* p = lds + n * rowStride + kBase + 8 * half;
    FragU f;
    f.h[0] = *(const ushort8*)p;
    f.h[1] = *(const ushort8*)(p + 16);
    return f.v;
}

// ---------------------------------------------------------------------------
// Tensor Data Mover: load one 2D bf16 tile [tile_rows x tile_elems_x] from a
// row-major matrix (row stride = stride_elems, elements) into LDS at
// lds_byte_addr, with LDS padding of (pad_amt_code+1) DWORDs inserted every
// 2^(pad_int_code+1) DWORDs (ISA D# §8).  Issued once per wave; EXEC ignored.
// ---------------------------------------------------------------------------
__device__ __forceinline__ void tdm_load_tile_2d(
    const unsigned short* gsrc, unsigned lds_byte_addr,
    unsigned tile_elems_x, unsigned tile_rows, unsigned stride_elems,
    unsigned pad_int_code, unsigned pad_amt_code) {
    unsigned long long ga = (unsigned long long)(size_t)gsrc;
    uint32x4 g0;
    g0[0] = 1u;                                        // count=1, user desc
    g0[1] = lds_byte_addr;                             // lds_addr
    g0[2] = (unsigned)ga;                              // global_addr[31:0]
    g0[3] = (unsigned)((ga >> 32) & 0x1FFFFFFu)        // global_addr[56:32]
            | (2u << 30);                              // type = 2 ("image")
    uint32x8 g1;
    g1[0] = (1u << 16)                                 // data_size = 2 bytes
          | (1u << 20)                                 // pad_enable
          | (pad_int_code << 22) | (pad_amt_code << 25);
    g1[1] = (stride_elems & 0xFFFFu) << 16;            // tensor_dim0[15:0]
    g1[2] = (stride_elems >> 16)                       // tensor_dim0[31:16]
          | ((tile_rows & 0xFFFFu) << 16);             // tensor_dim1[15:0]
    g1[3] = (tile_rows >> 16)                          // tensor_dim1[31:16]
          | ((tile_elems_x & 0xFFFFu) << 16);          // tile_dim0
    g1[4] = tile_rows & 0xFFFFu;                       // tile_dim1 (tile_dim2=0)
    g1[5] = stride_elems;                              // tensor_dim0_stride[31:0]
    g1[6] = 0;                                         // stride0[47:32], stride1 lo
    g1[7] = 0;                                         // stride1 hi
    asm volatile("tensor_load_to_lds %0, %1" :: "s"(g0), "s"(g1) : "memory");
}

__device__ __forceinline__ unsigned lds_addr_of(const void* p) {
    // Flat LDS aperture keeps the LDS byte offset in addr[31:0] (ISA 10.2).
    return (unsigned)(size_t)p;
}

// ---------------------------------------------------------------------------
// Weight convert+transpose: Wt[n*K+k] = bf16(W[k*N+n])
// ---------------------------------------------------------------------------
__global__ __launch_bounds__(256) void convert_transpose_kernel(
    const float* __restrict__ W, unsigned short* __restrict__ Wt, int K, int N) {
    size_t id = (size_t)blockIdx.x * 256 + threadIdx.x;
    if (id >= (size_t)K * N) return;
    int k = (int)(id / N);
    int n = (int)(id % N);
    Wt[(size_t)n * K + k] = f2bf(W[id]);
}

// ---------------------------------------------------------------------------
// LayerNorm: fp32 row [1024] -> bf16 row [1024]
// ---------------------------------------------------------------------------
__global__ __launch_bounds__(256) void ln_kernel(
    const float* __restrict__ x, const float* __restrict__ scale,
    const float* __restrict__ shift, unsigned short* __restrict__ out) {
    const int row = blockIdx.x;
    const float* xr = x + (size_t)row * 1024;
    const int tid = threadIdx.x;
    const int wave = tid >> 5, lane = tid & 31;

    float v[4];
    float s = 0.f;
    for (int i = 0; i < 4; ++i) { v[i] = xr[tid + i * 256]; s += v[i]; }
    for (int m = 16; m >= 1; m >>= 1) s += __shfl_xor(s, m, 32);
    __shared__ float red1[8], red2[8];
    if (lane == 0) red1[wave] = s;
    __syncthreads();
    float tot = 0.f;
    for (int i = 0; i < 8; ++i) tot += red1[i];
    const float mean = tot * (1.0f / 1024.0f);

    float vs = 0.f;
    for (int i = 0; i < 4; ++i) { float d = v[i] - mean; vs += d * d; }
    for (int m = 16; m >= 1; m >>= 1) vs += __shfl_xor(vs, m, 32);
    if (lane == 0) red2[wave] = vs;
    __syncthreads();
    float vtot = 0.f;
    for (int i = 0; i < 8; ++i) vtot += red2[i];
    const float rstd = rsqrtf(vtot * (1.0f / 1024.0f) + 1e-5f);

    unsigned short* orow = out + (size_t)row * 1024;
    for (int i = 0; i < 4; ++i) {
        int c = tid + i * 256;
        orow[c] = f2bf(scale[c] * (v[i] - mean) * rstd + shift[c]);
    }
}

// ---------------------------------------------------------------------------
// WMMA GEMM: C[M,N] = A[M,K](bf16) x Wt[N,K](bf16)^T, tile 128x128x32.
// Tiles staged to LDS by the Tensor Data Mover (one descriptor per tile).
// MODE 0: bf16 out, head-split [b,h,t,d] layout (QKV, N==1024)
// MODE 1: fp32 out = acc + bias[n] + res[m,n]
// MODE 2: bf16 out = gelu_exact(acc + bias[n])
// ---------------------------------------------------------------------------
template <int MODE>
__global__ __launch_bounds__(256) void gemm_kernel(
    const unsigned short* __restrict__ A, const unsigned short* __restrict__ Wt,
    const float* __restrict__ bias, const float* __restrict__ res,
    void* __restrict__ outp, int M, int N, int K) {
    constexpr int PAD = 40;   // 32 elems (16 DW) + TDM pad of 4 DW per row
    __shared__ __align__(16) unsigned short As[128 * PAD];
    __shared__ __align__(16) unsigned short Bs[128 * PAD];

    const int tid = threadIdx.x;
    const int wave = tid >> 5, lane = tid & 31;
    const int waveM = wave & 1, waveN = wave >> 1;   // 2 x 4 wave grid
    const int m0 = blockIdx.y * 128;
    const int n0 = blockIdx.x * 128;
    const unsigned ldsA = lds_addr_of(As);
    const unsigned ldsB = lds_addr_of(Bs);

    v8f acc[4][2];
    for (int i = 0; i < 4; ++i)
        for (int j = 0; j < 2; ++j)
            for (int e = 0; e < 8; ++e) acc[i][j][e] = 0.f;

    for (int k0 = 0; k0 < K; k0 += 32) {
        __syncthreads();
        if (tid < 32) {
            // pad: every 16 DWORDs (code 3) insert 4 DWORDs (code 3) -> 40-ushort rows
            tdm_load_tile_2d(A  + (size_t)m0 * K + k0, ldsA, 32, 128, (unsigned)K, 3, 3);
            tdm_load_tile_2d(Wt + (size_t)n0 * K + k0, ldsB, 32, 128, (unsigned)K, 3, 3);
            __builtin_amdgcn_s_wait_tensorcnt(0);
        }
        __syncthreads();

        v16bf a[4], b[2];
        for (int i = 0; i < 4; ++i) a[i] = load_afrag(As, PAD, waveM * 64 + i * 16, 0);
        for (int j = 0; j < 2; ++j) b[j] = load_bfrag(Bs, PAD, waveN * 32 + j * 16, 0);
        for (int i = 0; i < 4; ++i)
            for (int j = 0; j < 2; ++j)
                acc[i][j] = wmma_bf16(a[i], b[j], acc[i][j]);
    }

    const int half = lane >> 4;
    const int ln = lane & 15;
    for (int i = 0; i < 4; ++i) {
        for (int j = 0; j < 2; ++j) {
            const int gn = n0 + waveN * 32 + j * 16 + ln;
            for (int r = 0; r < 8; ++r) {
                const int gm = m0 + waveM * 64 + i * 16 + r + 8 * half;
                float v = acc[i][j][r];
                if constexpr (MODE == 0) {
                    int bb = gm >> 11, t = gm & 2047;
                    int h = gn >> 6, d = gn & 63;
                    ((unsigned short*)outp)[(((size_t)(bb * 16 + h) * 2048 + t) * 64) + d] =
                        f2bf(v);
                } else if constexpr (MODE == 1) {
                    ((float*)outp)[(size_t)gm * N + gn] =
                        v + bias[gn] + res[(size_t)gm * N + gn];
                } else {
                    float t = v + bias[gn];
                    float g = 0.5f * t * (1.f + erff(t * 0.70710678118654752f));
                    ((unsigned short*)outp)[(size_t)gm * N + gn] = f2bf(g);
                }
            }
        }
    }
}

// ---------------------------------------------------------------------------
// Flash attention (causal), per (b,h). Q-tile 128, KV-tile 64, D=64, T=2048.
// Q/K tiles staged by TDM; V transposed manually (TDM can't transpose).
// Q/K/V in bf16 [b,h,t,d]; ctx out bf16 [b,t,h*64+d] (= [token,1024]).
// ---------------------------------------------------------------------------
__global__ __launch_bounds__(256) void attn_kernel(
    const unsigned short* __restrict__ Qg, const unsigned short* __restrict__ Kg,
    const unsigned short* __restrict__ Vg, unsigned short* __restrict__ ctx) {
    constexpr int SPAD = 72;   // 64 elems (32 DW) + TDM pad of 4 DW per row
    __shared__ __align__(16) unsigned short Qs[128 * SPAD];
    __shared__ __align__(16) unsigned short Ks[64 * SPAD];
    __shared__ __align__(16) unsigned short Vts[64 * SPAD];   // [d][key]
    __shared__ __align__(16) unsigned short Ps[8 * 16 * SPAD]; // per-wave P staging

    const int tid = threadIdx.x;
    const int wave = tid >> 5, lane = tid & 31;
    const int half = lane >> 4, ln = lane & 15;
    const int bh = blockIdx.y;                 // 64 (b,h) pairs
    const int q0 = blockIdx.x * 128;
    const size_t base = (size_t)bh * 2048 * 64;

    // Q tile via TDM: 128 rows x 64 elems, pad every 32 DW (code 4) by 4 DW (code 3)
    if (tid < 32) {
        tdm_load_tile_2d(Qg + base + (size_t)q0 * 64, lds_addr_of(Qs), 64, 128, 64, 4, 3);
        __builtin_amdgcn_s_wait_tensorcnt(0);
    }
    __syncthreads();

    // Q A-fragments for this wave's 16 rows (invariant over KV tiles)
    v16bf qa[2];
    for (int s = 0; s < 2; ++s) qa[s] = load_afrag(Qs, SPAD, wave * 16, 32 * s);

    v8f o[4];
    for (int nd = 0; nd < 4; ++nd)
        for (int e = 0; e < 8; ++e) o[nd][e] = 0.f;
    float mrow[8], lrow[8];
    for (int r = 0; r < 8; ++r) { mrow[r] = -3.0e38f; lrow[r] = 0.f; }

    const int nkv = 2 * blockIdx.x + 2;   // KV tiles of 64 keys, up to diagonal
    for (int kt = 0; kt < nkv; ++kt) {
        const int k0 = kt * 64;
        __syncthreads();
        // K tile via TDM (overlaps with manual V transpose below)
        if (tid < 32) {
            tdm_load_tile_2d(Kg + base + (size_t)k0 * 64, lds_addr_of(Ks), 64, 64, 64, 4, 3);
        }
        // V tile transposed into LDS: Vts[d][key]
        for (int c = tid; c < 512; c += 256) {
            int row = c >> 3, col = (c & 7) * 8;
            ushort8 vv = *(const ushort8*)&Vg[base + (size_t)(k0 + row) * 64 + col];
            for (int e = 0; e < 8; ++e) Vts[(col + e) * SPAD + row] = vv[e];
        }
        if (tid < 32) __builtin_amdgcn_s_wait_tensorcnt(0);
        __syncthreads();

        // S = Q K^T : 4 key-frags x (2 k-steps over d)
        v8f s8[4];
        for (int j = 0; j < 4; ++j) {
            v8f z;
            for (int e = 0; e < 8; ++e) z[e] = 0.f;
            z = wmma_bf16(qa[0], load_bfrag(Ks, SPAD, j * 16, 0), z);
            z = wmma_bf16(qa[1], load_bfrag(Ks, SPAD, j * 16, 32), z);
            s8[j] = z;
        }

        // scale + causal mask + row max
        float tmax[8];
        for (int r = 0; r < 8; ++r) tmax[r] = -3.0e38f;
        for (int j = 0; j < 4; ++j) {
            const int kg = k0 + j * 16 + ln;
            for (int r = 0; r < 8; ++r) {
                const int qg = q0 + wave * 16 + r + 8 * half;
                float val = s8[j][r] * 0.125f;           // 1/sqrt(64)
                if (kg > qg) val = -1.0e30f;
                s8[j][r] = val;
                tmax[r] = fmaxf(tmax[r], val);
            }
        }
        for (int r = 0; r < 8; ++r)
            for (int m = 1; m <= 8; m <<= 1)
                tmax[r] = fmaxf(tmax[r], __shfl_xor(tmax[r], m, 32));

        float fac[8];
        for (int r = 0; r < 8; ++r) {
            float mnew = fmaxf(mrow[r], tmax[r]);
            fac[r] = __expf(mrow[r] - mnew);
            mrow[r] = mnew;
        }

        float tsum[8];
        for (int r = 0; r < 8; ++r) tsum[r] = 0.f;
        for (int j = 0; j < 4; ++j)
            for (int r = 0; r < 8; ++r) {
                float p = __expf(s8[j][r] - mrow[r]);
                s8[j][r] = p;
                tsum[r] += p;
            }
        for (int r = 0; r < 8; ++r)
            for (int m = 1; m <= 8; m <<= 1)
                tsum[r] += __shfl_xor(tsum[r], m, 32);
        for (int r = 0; r < 8; ++r) lrow[r] = lrow[r] * fac[r] + tsum[r];

        // rescale O accumulators
        for (int nd = 0; nd < 4; ++nd)
            for (int r = 0; r < 8; ++r) o[nd][r] *= fac[r];

        // stage P (C-layout -> LDS row-major) for A-fragment reload
        unsigned short* pw = &Ps[wave * 16 * SPAD];
        for (int j = 0; j < 4; ++j)
            for (int r = 0; r < 8; ++r)
                pw[(r + 8 * half) * SPAD + j * 16 + ln] = f2bf(s8[j][r]);
        asm volatile("s_wait_dscnt 0" ::: "memory");

        v16bf pf[2];
        for (int st = 0; st < 2; ++st) pf[st] = load_afrag(pw, SPAD, 0, 32 * st);
        for (int nd = 0; nd < 4; ++nd)
            for (int st = 0; st < 2; ++st)
                o[nd] = wmma_bf16(pf[st], load_bfrag(Vts, SPAD, nd * 16, 32 * st), o[nd]);
    }

    // normalize and write ctx [token, h*64+d] as bf16
    const int b = bh >> 4, h = bh & 15;
    for (int nd = 0; nd < 4; ++nd) {
        const int col = h * 64 + nd * 16 + ln;
        for (int r = 0; r < 8; ++r) {
            const int trow = q0 + wave * 16 + r + 8 * half;
            const float val = o[nd][r] / lrow[r];
            ctx[(size_t)(b * 2048 + trow) * 1024 + col] = f2bf(val);
        }
    }
}

// ---------------------------------------------------------------------------
// Launcher
// ---------------------------------------------------------------------------
extern "C" void kernel_launch(void* const* d_in, const int* in_sizes, int n_in,
                              void* d_out, int out_size, void* d_ws, size_t ws_size,
                              hipStream_t stream) {
    const float* x   = (const float*)d_in[0];
    const float* Wq  = (const float*)d_in[1];
    const float* Wk  = (const float*)d_in[2];
    const float* Wv  = (const float*)d_in[3];
    const float* Wo  = (const float*)d_in[4];
    const float* bo  = (const float*)d_in[5];
    const float* W1  = (const float*)d_in[6];
    const float* b1  = (const float*)d_in[7];
    const float* W2  = (const float*)d_in[8];
    const float* b2  = (const float*)d_in[9];
    const float* ln1s = (const float*)d_in[10];
    const float* ln1b = (const float*)d_in[11];
    const float* ln2s = (const float*)d_in[12];
    const float* ln2b = (const float*)d_in[13];

    const int M = 4 * 2048;      // 8192 tokens
    char* ws = (char*)d_ws;
    size_t off = 0;
    auto alloc = [&](size_t bytes) { char* p = ws + off; off += bytes; return p; };

    unsigned short* WtQ = (unsigned short*)alloc((size_t)1024 * 1024 * 2);
    unsigned short* WtK = (unsigned short*)alloc((size_t)1024 * 1024 * 2);
    unsigned short* WtV = (unsigned short*)alloc((size_t)1024 * 1024 * 2);
    unsigned short* WtO = (unsigned short*)alloc((size_t)1024 * 1024 * 2);
    unsigned short* Wt1 = (unsigned short*)alloc((size_t)4096 * 1024 * 2);
    unsigned short* Wt2 = (unsigned short*)alloc((size_t)1024 * 4096 * 2);
    unsigned short* xln = (unsigned short*)alloc((size_t)M * 1024 * 2);  // reused as h2
    unsigned short* Qb  = (unsigned short*)alloc((size_t)M * 1024 * 2);
    unsigned short* Kb  = (unsigned short*)alloc((size_t)M * 1024 * 2);
    unsigned short* Vb  = (unsigned short*)alloc((size_t)M * 1024 * 2);
    unsigned short* ctx = (unsigned short*)alloc((size_t)M * 1024 * 2);
    float*          x1  = (float*)alloc((size_t)M * 1024 * 4);
    unsigned short* ff1 = (unsigned short*)alloc((size_t)M * 4096 * 2);
    (void)ws_size;

    // 1) weights -> bf16 transposed
    {
        int n1 = (1024 * 1024 + 255) / 256;
        convert_transpose_kernel<<<n1, 256, 0, stream>>>(Wq, WtQ, 1024, 1024);
        convert_transpose_kernel<<<n1, 256, 0, stream>>>(Wk, WtK, 1024, 1024);
        convert_transpose_kernel<<<n1, 256, 0, stream>>>(Wv, WtV, 1024, 1024);
        convert_transpose_kernel<<<n1, 256, 0, stream>>>(Wo, WtO, 1024, 1024);
        int n2 = (1024 * 4096 + 255) / 256;
        convert_transpose_kernel<<<n2, 256, 0, stream>>>(W1, Wt1, 1024, 4096);
        convert_transpose_kernel<<<n2, 256, 0, stream>>>(W2, Wt2, 4096, 1024);
    }

    // 2) LN1
    ln_kernel<<<M, 256, 0, stream>>>(x, ln1s, ln1b, xln);

    // 3) Q,K,V projections (head-split bf16 output)
    dim3 gqkv(1024 / 128, M / 128);
    gemm_kernel<0><<<gqkv, 256, 0, stream>>>(xln, WtQ, nullptr, nullptr, Qb, M, 1024, 1024);
    gemm_kernel<0><<<gqkv, 256, 0, stream>>>(xln, WtK, nullptr, nullptr, Kb, M, 1024, 1024);
    gemm_kernel<0><<<gqkv, 256, 0, stream>>>(xln, WtV, nullptr, nullptr, Vb, M, 1024, 1024);

    // 4) causal flash attention
    attn_kernel<<<dim3(2048 / 128, 64), 256, 0, stream>>>(Qb, Kb, Vb, ctx);

    // 5) output projection + bias + residual -> x1 (fp32)
    gemm_kernel<1><<<dim3(1024 / 128, M / 128), 256, 0, stream>>>(
        ctx, WtO, bo, x, x1, M, 1024, 1024);

    // 6) LN2 (reuse xln buffer)
    ln_kernel<<<M, 256, 0, stream>>>(x1, ln2s, ln2b, xln);

    // 7) FF1 + exact GELU -> bf16
    gemm_kernel<2><<<dim3(4096 / 128, M / 128), 256, 0, stream>>>(
        xln, Wt1, b1, nullptr, ff1, M, 4096, 1024);

    // 8) FF2 + bias + residual -> d_out (fp32)
    gemm_kernel<1><<<dim3(1024 / 128, M / 128), 256, 0, stream>>>(
        ff1, Wt2, b2, x1, (float*)d_out, M, 1024, 4096);
}
